// Mamba_GCL_7567732375773
// MI455X (gfx1250) — matrix operand order repaired and checked by reading, not compiled, not run on hardware
//
#include <hip/hip_runtime.h>
#include <hip/hip_bf16.h>
#include <stdint.h>

#define HDIM 128
#define DSQ  64
#define KP1  288     // 258 (2H+EA) padded to 9 K-chunks of 32 (576B row stride)
#define CH_L 100     // scan chunk length

typedef __bf16          v16bf __attribute__((ext_vector_type(16)));
typedef float           v8f   __attribute__((ext_vector_type(8)));
typedef unsigned short  v16us __attribute__((ext_vector_type(16)));
typedef unsigned int    v4u   __attribute__((ext_vector_type(4)));
typedef int             v4i   __attribute__((ext_vector_type(4)));
typedef int             v8i   __attribute__((ext_vector_type(8)));

__device__ __forceinline__ unsigned short f2bf(float f) {
  union { float f; unsigned int u; } c; c.f = f;
  unsigned int u = c.u;
  unsigned int r = u + 0x7FFFu + ((u >> 16) & 1u);  // round-to-nearest-even
  return (unsigned short)(r >> 16);
}
__device__ __forceinline__ float sigf(float x)  { return 1.0f / (1.0f + __expf(-x)); }
__device__ __forceinline__ float siluf(float x) { return x * sigf(x); }

__device__ __forceinline__ v8f wmma_bf16(v16bf a, v16bf b, v8f c) {
  return __builtin_amdgcn_wmma_f32_16x16x32_bf16(false, a, false, b, (short)0, c, false, false);
}

// A-fragment (16x32 bf16, MxK) from LDS row-major [16][kpad] halves.
// ISA layout: lane<16 -> M=lane, halves 0..7 = K 8kh.., 8..15 = K 16+8kh..
__device__ __forceinline__ v16bf load_a(const unsigned short* Al, int kpad, int kc, int lane) {
  int m = lane & 15, kh = lane >> 4;
  const unsigned short* p = Al + m * kpad + kc * 32 + 8 * kh;
  v16us u;
#pragma unroll
  for (int j = 0; j < 8; ++j) { u[j] = p[j]; u[8 + j] = p[16 + j]; }
  return __builtin_bit_cast(v16bf, u);
}
// B-fragment from prepped blob: [kc][nt][lane][16 halves], K = kc*32 + (j<8 ? 8*(lane>>4)+j : 16+8*(lane>>4)+j-8), N = nt*16 + (lane&15)
__device__ __forceinline__ v16bf load_b(const unsigned short* Bf, int ntc, int kc, int nt, int lane) {
  const v16us* p = (const v16us*)(Bf + (((size_t)kc * ntc + nt) * 32 + lane) * 16);
  v16us u = *p;
  return __builtin_bit_cast(v16bf, u);
}

// ---------------- prep kernels ----------------
__global__ __launch_bounds__(256) void prep_frags(const float* __restrict__ W,
                                                  unsigned short* __restrict__ dst,
                                                  int kc_count, int nt_count, int Kreal, int Nreal) {
  int total = kc_count * nt_count * 512;
  for (int i = blockIdx.x * blockDim.x + threadIdx.x; i < total; i += gridDim.x * blockDim.x) {
    int j    = i & 15;
    int lane = (i >> 4) & 31;
    int rest = i >> 9;
    int nt   = rest % nt_count;
    int kc   = rest / nt_count;
    int K    = kc * 32 + ((j < 8) ? (lane >> 4) * 8 + j : 16 + (lane >> 4) * 8 + (j - 8));
    int Ncol = nt * 16 + (lane & 15);
    float v = (K < Kreal && Ncol < Nreal) ? W[(size_t)K * Nreal + Ncol] : 0.0f;
    dst[i] = f2bf(v);
  }
}

__global__ __launch_bounds__(256) void prep_misc(const float* __restrict__ h,
                                                 const float* __restrict__ A_log,
                                                 unsigned short* __restrict__ hb,
                                                 float* __restrict__ Aneg,
                                                 int n_h, int n_a) {
  int i0 = blockIdx.x * blockDim.x + threadIdx.x, st = gridDim.x * blockDim.x;
  for (int i = i0; i < n_h; i += st) hb[i] = f2bf(h[i]);
  for (int i = i0; i < n_a; i += st) Aneg[i] = -__expf(A_log[i]);
}

// ---------------- edge MLP + scatter ----------------
__global__ __launch_bounds__(128) void k_edge(
    const unsigned short* __restrict__ hb, const int* __restrict__ eidx,
    const float* __restrict__ eattr,
    const unsigned short* __restrict__ w1f, const float* __restrict__ b1,
    const unsigned short* __restrict__ w2f, const float* __restrict__ b2,
    const float* __restrict__ infw, const float* __restrict__ infb,
    float* __restrict__ mi, float* __restrict__ agg, int Etot, int Ntot) {
  __shared__ unsigned short Al[16 * KP1];
  __shared__ unsigned short A2[16 * HDIM];
  __shared__ float Ml[16 * HDIM];
  __shared__ float El[16];
  __shared__ int rI[16], cI[16];
  int tid = threadIdx.x, lane = tid & 31, wv = tid >> 5;
  int e0 = blockIdx.x * 16;
  if (tid < 16) {
    int ge = e0 + tid;
    rI[tid] = (ge < Etot) ? eidx[ge] : 0;
    cI[tid] = (ge < Etot) ? eidx[Etot + ge] : 0;
  }

#if __has_builtin(__builtin_amdgcn_tensor_load_to_lds)
  // TDM gather: DMA 16 rows of hb (h[row[e]] and h[col[e]], 256B each) into the
  // 16x288-half LDS A-tile. pad_interval=5 (64 DW = one row), pad_amount=79
  // (80 DW = 320B) -> effective LDS row stride 576B == KP1 halves.
  if (wv == 0) {
    int vr = 0, vc = 0;
    {
      int ge = e0 + lane;
      if (lane < 16 && ge < Etot) { vr = eidx[ge]; vc = eidx[Etot + ge]; }
    }
    v4i g2r, g3r, g2c, g3c;
#pragma unroll
    for (int k = 0; k < 4; ++k) {
      unsigned a0 = (unsigned)__builtin_amdgcn_readlane(vr, 2 * k + 0) & 0xFFFFu;
      unsigned a1 = (unsigned)__builtin_amdgcn_readlane(vr, 2 * k + 1) & 0xFFFFu;
      unsigned a2 = (unsigned)__builtin_amdgcn_readlane(vr, 8 + 2 * k + 0) & 0xFFFFu;
      unsigned a3 = (unsigned)__builtin_amdgcn_readlane(vr, 8 + 2 * k + 1) & 0xFFFFu;
      g2r[k] = (int)(a0 | (a1 << 16));
      g3r[k] = (int)(a2 | (a3 << 16));
      unsigned b0 = (unsigned)__builtin_amdgcn_readlane(vc, 2 * k + 0) & 0xFFFFu;
      unsigned b1_ = (unsigned)__builtin_amdgcn_readlane(vc, 2 * k + 1) & 0xFFFFu;
      unsigned b2_ = (unsigned)__builtin_amdgcn_readlane(vc, 8 + 2 * k + 0) & 0xFFFFu;
      unsigned b3 = (unsigned)__builtin_amdgcn_readlane(vc, 8 + 2 * k + 1) & 0xFFFFu;
      g2c[k] = (int)(b0 | (b1_ << 16));
      g3c[k] = (int)(b2_ | (b3 << 16));
    }
    // D# group1: wg_mask=0 | data_size=1(2B) | pad_en | pad_interval=5 | pad_amount=79
    v8i g1 = { (int)0x9F510000u,
               (int)(128u << 16),                                   // tensor_dim0 = 128
               (int)(((unsigned)Ntot & 0xFFFFu) << 16),             // tensor_dim1 lo
               (int)((((unsigned)Ntot >> 16) & 0xFFFFu) | (128u << 16)), // tensor_dim1 hi | tile_dim0=128
               16,                                                  // tile_dim1 = #indices
               128,                                                 // tensor_dim0_stride
               0, 0 };
    unsigned lds_base = (unsigned)(uintptr_t)(void*)Al;
    unsigned long long ga = (unsigned long long)(uintptr_t)hb;
    v4u g0;
    g0[0] = 0x80000001u;  // count=1 | gather_mode | 16-bit indices
    g0[1] = lds_base;     // dest: Al[e][0:128]
    g0[2] = (unsigned)ga;
    g0[3] = (unsigned)((ga >> 32) & 0x01FFFFFFu) | 0x80000000u;  // addr hi | type=2
#if defined(__clang_major__) && __clang_major__ >= 23
    v8i gz = {};
    __builtin_amdgcn_tensor_load_to_lds(g0, g1, g2r, g3r, gz, 0);
    g0[1] = lds_base + 256;  // dest: Al[e][128:256]
    __builtin_amdgcn_tensor_load_to_lds(g0, g1, g2c, g3c, gz, 0);
#else
    __builtin_amdgcn_tensor_load_to_lds(g0, g1, g2r, g3r, 0);
    g0[1] = lds_base + 256;  // dest: Al[e][128:256]
    __builtin_amdgcn_tensor_load_to_lds(g0, g1, g2c, g3c, 0);
#endif
    __builtin_amdgcn_s_wait_tensorcnt(0);
  }
  // VALU fills only the edge_attr + pad columns (fb 256..287)
  for (int c = tid; c < 16 * 8; c += 128) {
    int e = c >> 3, fb = 256 + ((c & 7) << 2), ge = e0 + e;
    unsigned short o[4] = {0, 0, 0, 0};
    if (ge < Etot && fb == 256) {
      o[0] = f2bf(eattr[(size_t)ge * 2 + 0]);
      o[1] = f2bf(eattr[(size_t)ge * 2 + 1]);
    }
    unsigned short* d = Al + e * KP1 + fb;
    d[0] = o[0]; d[1] = o[1]; d[2] = o[2]; d[3] = o[3];
  }
#else
  __syncthreads();
  for (int c = tid; c < 16 * 72; c += 128) {
    int e = c / 72, fb = (c % 72) * 4, ge = e0 + e;
    unsigned short o[4] = {0, 0, 0, 0};
    if (ge < Etot) {
      if (fb < 256) {
        int node = (fb < 128) ? rI[e] : cI[e];
        int f = (fb < 128) ? fb : fb - 128;
        const unsigned short* s = hb + (size_t)node * HDIM + f;
        o[0] = s[0]; o[1] = s[1]; o[2] = s[2]; o[3] = s[3];
      } else if (fb == 256) {
        o[0] = f2bf(eattr[(size_t)ge * 2 + 0]);
        o[1] = f2bf(eattr[(size_t)ge * 2 + 1]);
      }
    }
    unsigned short* d = Al + e * KP1 + fb;
    d[0] = o[0]; d[1] = o[1]; d[2] = o[2]; d[3] = o[3];
  }
#endif
  __syncthreads();
  int kh = lane >> 4, nl = lane & 15;
  // layer1: (16x288)@(288x128), bias + silu -> A2 (bf16)
  {
    v8f a0 = {}, a1 = {};
#pragma unroll
    for (int kc = 0; kc < 9; ++kc) {
      v16bf a = load_a(Al, KP1, kc, lane);
      a0 = wmma_bf16(a, load_b(w1f, 8, kc, 2 * wv + 0, lane), a0);
      a1 = wmma_bf16(a, load_b(w1f, 8, kc, 2 * wv + 1, lane), a1);
    }
#pragma unroll
    for (int t = 0; t < 2; ++t) {
      int n = (2 * wv + t) * 16 + nl; float bb = b1[n];
      v8f acc = t ? a1 : a0;
#pragma unroll
      for (int r = 0; r < 8; ++r) A2[(r + 8 * kh) * HDIM + n] = f2bf(siluf(acc[r] + bb));
    }
  }
  __syncthreads();
  // layer2: (16x128)@(128x128), bias + silu -> Ml (f32) == mij
  {
    v8f a0 = {}, a1 = {};
#pragma unroll
    for (int kc = 0; kc < 4; ++kc) {
      v16bf a = load_a(A2, HDIM, kc, lane);
      a0 = wmma_bf16(a, load_b(w2f, 8, kc, 2 * wv + 0, lane), a0);
      a1 = wmma_bf16(a, load_b(w2f, 8, kc, 2 * wv + 1, lane), a1);
    }
#pragma unroll
    for (int t = 0; t < 2; ++t) {
      int n = (2 * wv + t) * 16 + nl; float bb = b2[n];
      v8f acc = t ? a1 : a0;
#pragma unroll
      for (int r = 0; r < 8; ++r) Ml[(r + 8 * kh) * HDIM + n] = siluf(acc[r] + bb);
    }
  }
  __syncthreads();
  if (tid < 16) {
    float d = infb[0];
    for (int k = 0; k < HDIM; ++k) d += Ml[tid * HDIM + k] * infw[k];
    El[tid] = sigf(d);
  }
  __syncthreads();
  // scatter: mi[col] += mij*eij ; agg[row] += mij
  for (int i = tid; i < 16 * HDIM; i += 128) {
    int r = i >> 7, f = i & 127, ge = e0 + r;
    if (ge < Etot) {
      float m = Ml[i];
      atomicAdd(&mi[(size_t)cI[r] * HDIM + f], m * El[r]);
      atomicAdd(&agg[(size_t)rI[r] * HDIM + f], m);
    }
  }
}

// ---------------- node MLPs + in_proj ----------------
__global__ __launch_bounds__(128) void k_node(
    const unsigned short* __restrict__ hb, const float* __restrict__ h,
    const float* __restrict__ agg, const float* __restrict__ mi,
    const unsigned short* __restrict__ nw1f, const float* __restrict__ nb1,
    const unsigned short* __restrict__ nw2f, const float* __restrict__ nb2,
    const unsigned short* __restrict__ inpf,
    float* __restrict__ xraw, float* __restrict__ zbuf, int Ntot) {
  __shared__ unsigned short Ab[16 * 256];
  __shared__ unsigned short Tb[16 * HDIM];
  __shared__ unsigned short Hb[16 * HDIM];
  int tid = threadIdx.x, lane = tid & 31, wv = tid >> 5;
  int n0 = blockIdx.x * 16;
  int kh = lane >> 4, nl = lane & 15;
  // concat(h, agg)
  for (int i = tid; i < 16 * 256; i += 128) {
    int r = i >> 8, f = i & 255, node = n0 + r;
    unsigned short v = 0;
    if (node < Ntot)
      v = (f < 128) ? hb[(size_t)node * HDIM + f] : f2bf(agg[(size_t)node * HDIM + (f - 128)]);
    Ab[i] = v;
  }
  __syncthreads();
  // mlp1 layer1 (256->128, silu)
  {
    v8f a0 = {}, a1 = {};
#pragma unroll
    for (int kc = 0; kc < 8; ++kc) {
      v16bf a = load_a(Ab, 256, kc, lane);
      a0 = wmma_bf16(a, load_b(nw1f, 8, kc, 2 * wv + 0, lane), a0);
      a1 = wmma_bf16(a, load_b(nw1f, 8, kc, 2 * wv + 1, lane), a1);
    }
#pragma unroll
    for (int t = 0; t < 2; ++t) {
      int n = (2 * wv + t) * 16 + nl; float bb = nb1[n]; v8f acc = t ? a1 : a0;
#pragma unroll
      for (int r = 0; r < 8; ++r) Tb[(r + 8 * kh) * HDIM + n] = f2bf(siluf(acc[r] + bb));
    }
  }
  __syncthreads();
  // mlp1 layer2 + residual -> h1 into Ab[:,128:256]; mi -> Ab[:,0:128]
  {
    v8f a0 = {}, a1 = {};
#pragma unroll
    for (int kc = 0; kc < 4; ++kc) {
      v16bf a = load_a(Tb, HDIM, kc, lane);
      a0 = wmma_bf16(a, load_b(nw2f, 8, kc, 2 * wv + 0, lane), a0);
      a1 = wmma_bf16(a, load_b(nw2f, 8, kc, 2 * wv + 1, lane), a1);
    }
#pragma unroll
    for (int t = 0; t < 2; ++t) {
      int n = (2 * wv + t) * 16 + nl; float bb = nb2[n]; v8f acc = t ? a1 : a0;
#pragma unroll
      for (int r = 0; r < 8; ++r) {
        int row = r + 8 * kh, node = n0 + row;
        float res = (node < Ntot) ? h[(size_t)node * HDIM + n] : 0.f;
        Ab[row * 256 + 128 + n] = f2bf(acc[r] + bb + res);
      }
    }
    for (int i = tid; i < 16 * HDIM; i += 128) {
      int r = i >> 7, f = i & 127, node = n0 + r;
      Ab[r * 256 + f] = (node < Ntot) ? f2bf(mi[(size_t)node * HDIM + f]) : (unsigned short)0;
    }
  }
  __syncthreads();
  // mlp2 layer1
  {
    v8f a0 = {}, a1 = {};
#pragma unroll
    for (int kc = 0; kc < 8; ++kc) {
      v16bf a = load_a(Ab, 256, kc, lane);
      a0 = wmma_bf16(a, load_b(nw1f, 8, kc, 2 * wv + 0, lane), a0);
      a1 = wmma_bf16(a, load_b(nw1f, 8, kc, 2 * wv + 1, lane), a1);
    }
#pragma unroll
    for (int t = 0; t < 2; ++t) {
      int n = (2 * wv + t) * 16 + nl; float bb = nb1[n]; v8f acc = t ? a1 : a0;
#pragma unroll
      for (int r = 0; r < 8; ++r) Tb[(r + 8 * kh) * HDIM + n] = f2bf(siluf(acc[r] + bb));
    }
  }
  __syncthreads();
  // mlp2 layer2 -> h2 (Hb)
  {
    v8f a0 = {}, a1 = {};
#pragma unroll
    for (int kc = 0; kc < 4; ++kc) {
      v16bf a = load_a(Tb, HDIM, kc, lane);
      a0 = wmma_bf16(a, load_b(nw2f, 8, kc, 2 * wv + 0, lane), a0);
      a1 = wmma_bf16(a, load_b(nw2f, 8, kc, 2 * wv + 1, lane), a1);
    }
#pragma unroll
    for (int t = 0; t < 2; ++t) {
      int n = (2 * wv + t) * 16 + nl; float bb = nb2[n]; v8f acc = t ? a1 : a0;
#pragma unroll
      for (int r = 0; r < 8; ++r) Hb[(r + 8 * kh) * HDIM + n] = f2bf(acc[r] + bb);
    }
  }
  __syncthreads();
  // in_proj: h2 (16x128) @ (128x256) -> x | z
  {
    v8f acc[4] = {{}, {}, {}, {}};
#pragma unroll
    for (int kc = 0; kc < 4; ++kc) {
      v16bf a = load_a(Hb, HDIM, kc, lane);
#pragma unroll
      for (int t = 0; t < 4; ++t) acc[t] = wmma_bf16(a, load_b(inpf, 16, kc, 4 * wv + t, lane), acc[t]);
    }
#pragma unroll
    for (int t = 0; t < 4; ++t) {
      int n = (4 * wv + t) * 16 + nl;
#pragma unroll
      for (int r = 0; r < 8; ++r) {
        int node = n0 + r + 8 * kh;
        if (node < Ntot) {
          float v = acc[t][r];
          if (n < 128) xraw[(size_t)node * HDIM + n] = v;
          else         zbuf[(size_t)node * HDIM + (n - 128)] = v;
        }
      }
    }
  }
}

// ---------------- causal conv + x_proj + delta ----------------
__global__ __launch_bounds__(128) void k_conv(
    const float* __restrict__ xraw, const float* __restrict__ convw, const float* __restrict__ convb,
    const unsigned short* __restrict__ xpf, const unsigned short* __restrict__ dtwf,
    const float* __restrict__ dtb,
    float* __restrict__ xc, float* __restrict__ Bm, float* __restrict__ Cm,
    float* __restrict__ delta, int Ntot) {
  __shared__ unsigned short Xb[16 * HDIM];
  __shared__ unsigned short Dt[16 * 32];
  int tid = threadIdx.x, lane = tid & 31, wv = tid >> 5;
  int n0 = blockIdx.x * 16;
  for (int i = tid; i < 16 * 32; i += 128) Dt[i] = 0;
  for (int i = tid; i < 16 * HDIM; i += 128) {
    int r = i >> 7, f = i & 127, node = n0 + r;
    float v = 0.f;
    if (node < Ntot) {
      v = convb[f];
#pragma unroll
      for (int k = 0; k < 4; ++k) {
        int idx = node - 3 + k;
        if (idx >= 0) v += convw[k * HDIM + f] * xraw[(size_t)idx * HDIM + f];
      }
      v = siluf(v);
      xc[(size_t)node * HDIM + f] = v;
    }
    Xb[i] = f2bf(v);
  }
  __syncthreads();
  int kh = lane >> 4, nl = lane & 15;
  // x_proj: (16x128)@(128x144pad) -> dt(8) | B(64) | C(64) | pad
  {
    v8f acc[3] = {{}, {}, {}};
#pragma unroll
    for (int kc = 0; kc < 4; ++kc) {
      v16bf a = load_a(Xb, HDIM, kc, lane);
      for (int t = 0; t < 3; ++t) {
        int nt = wv + 4 * t;
        if (nt < 9) acc[t] = wmma_bf16(a, load_b(xpf, 9, kc, nt, lane), acc[t]);
      }
    }
    for (int t = 0; t < 3; ++t) {
      int nt = wv + 4 * t;
      if (nt >= 9) break;
#pragma unroll
      for (int r = 0; r < 8; ++r) {
        int row = r + 8 * kh, node = n0 + row, col = nt * 16 + nl;
        float v = acc[t][r];
        if (col < 8)        Dt[row * 32 + col] = f2bf(v);
        else if (col < 72)  { if (node < Ntot) Bm[(size_t)node * DSQ + (col - 8)]  = v; }
        else if (col < 136) { if (node < Ntot) Cm[(size_t)node * DSQ + (col - 72)] = v; }
      }
    }
  }
  __syncthreads();
  // delta = softplus(dt @ dt_w + dt_b)   (K padded 8->32)
  {
    v8f a0 = {}, a1 = {};
    v16bf a = load_a(Dt, 32, 0, lane);
    a0 = wmma_bf16(a, load_b(dtwf, 8, 0, 2 * wv + 0, lane), a0);
    a1 = wmma_bf16(a, load_b(dtwf, 8, 0, 2 * wv + 1, lane), a1);
#pragma unroll
    for (int t = 0; t < 2; ++t) {
      int n = (2 * wv + t) * 16 + nl; float bb = dtb[n]; v8f acc = t ? a1 : a0;
#pragma unroll
      for (int r = 0; r < 8; ++r) {
        int node = n0 + r + 8 * kh;
        if (node < Ntot) {
          float v = acc[r] + bb;
          delta[(size_t)node * HDIM + n] = (v > 20.f) ? v : __logf(1.0f + __expf(v));
        }
      }
    }
  }
}

// ---------------- chunked SSM scan ----------------
__global__ __launch_bounds__(256) void k_scan1(
    const float* __restrict__ delta, const float* __restrict__ xc, const float* __restrict__ Bm,
    const float* __restrict__ Aneg, float* __restrict__ Slocal, float* __restrict__ dsum,
    int Ntot, int L) {
  int tid = threadIdx.x;
  int i = tid >> 1, jb = (tid & 1) * 32;
  int c = blockIdx.x, g0 = c * L;
  float s[32];
#pragma unroll
  for (int j = 0; j < 32; ++j) s[j] = 0.f;
  float ds = 0.f;
  for (int t = 0; t < L; ++t) {
    int g = g0 + t;
    if (g >= Ntot) break;
    float d  = delta[(size_t)g * HDIM + i];
    float dx = d * xc[(size_t)g * HDIM + i];
    ds += d;
#pragma unroll
    for (int j = 0; j < 32; ++j) {
      float a = __expf(d * Aneg[i * DSQ + jb + j]);
      s[j] = a * s[j] + dx * Bm[(size_t)g * DSQ + jb + j];
    }
  }
  float* o = Slocal + (size_t)c * HDIM * DSQ + (size_t)i * DSQ + jb;
#pragma unroll
  for (int j = 0; j < 32; ++j) o[j] = s[j];
  if ((tid & 1) == 0) dsum[(size_t)c * HDIM + i] = ds;
}

__global__ __launch_bounds__(256) void k_scan2(
    const float* __restrict__ Slocal, const float* __restrict__ dsum,
    const float* __restrict__ Aneg, float* __restrict__ Sinit, int NC) {
  int tid = threadIdx.x;
  int i = tid >> 1, jb = (tid & 1) * 32;
  float s[32];
#pragma unroll
  for (int j = 0; j < 32; ++j) s[j] = 0.f;
  float* o0 = Sinit + (size_t)i * DSQ + jb;
#pragma unroll
  for (int j = 0; j < 32; ++j) o0[j] = 0.f;
  for (int c = 1; c < NC; ++c) {
    float d = dsum[(size_t)(c - 1) * HDIM + i];
    const float* sl = Slocal + (size_t)(c - 1) * HDIM * DSQ + (size_t)i * DSQ + jb;
    float* o = Sinit + (size_t)c * HDIM * DSQ + (size_t)i * DSQ + jb;
#pragma unroll
    for (int j = 0; j < 32; ++j) {
      s[j] = __expf(d * Aneg[i * DSQ + jb + j]) * s[j] + sl[j];
      o[j] = s[j];
    }
  }
}

__global__ __launch_bounds__(256) void k_scan3(
    const float* __restrict__ delta, const float* __restrict__ xc, const float* __restrict__ Bm,
    const float* __restrict__ Cm, const float* __restrict__ Aneg,
    const float* __restrict__ Sinit, float* __restrict__ y, int Ntot, int L) {
  int tid = threadIdx.x;
  int i = tid >> 1, jb = (tid & 1) * 32;
  int c = blockIdx.x, g0 = c * L;
  float s[32];
  const float* si = Sinit + (size_t)c * HDIM * DSQ + (size_t)i * DSQ + jb;
#pragma unroll
  for (int j = 0; j < 32; ++j) s[j] = si[j];
  for (int t = 0; t < L; ++t) {
    int g = g0 + t;
    if (g >= Ntot) break;
    if (g + 1 < Ntot) {
      __builtin_prefetch(&Bm[(size_t)(g + 1) * DSQ + jb], 0, 1);
      __builtin_prefetch(&Cm[(size_t)(g + 1) * DSQ + jb], 0, 1);
    }
    float d  = delta[(size_t)g * HDIM + i];
    float dx = d * xc[(size_t)g * HDIM + i];
    float p = 0.f;
#pragma unroll
    for (int j = 0; j < 32; ++j) {
      float a = __expf(d * Aneg[i * DSQ + jb + j]);
      s[j] = a * s[j] + dx * Bm[(size_t)g * DSQ + jb + j];
      p += s[j] * Cm[(size_t)g * DSQ + jb + j];
    }
    float po = __shfl_xor(p, 1);
    if ((tid & 1) == 0) y[(size_t)g * HDIM + i] = p + po;
  }
}

// ---------------- gate + out_proj ----------------
__global__ __launch_bounds__(128) void k_final(
    const float* __restrict__ y, const float* __restrict__ xc, const float* __restrict__ zbuf,
    const float* __restrict__ Dv, const unsigned short* __restrict__ outf,
    float* __restrict__ out, int Ntot) {
  __shared__ unsigned short Yb[16 * HDIM];
  int tid = threadIdx.x, lane = tid & 31, wv = tid >> 5;
  int n0 = blockIdx.x * 16;
  for (int i = tid; i < 16 * HDIM; i += 128) {
    int r = i >> 7, f = i & 127, node = n0 + r;
    float v = 0.f;
    if (node < Ntot) {
      float yy = y[(size_t)node * HDIM + f] + Dv[f] * xc[(size_t)node * HDIM + f];
      v = yy * siluf(zbuf[(size_t)node * HDIM + f]);
    }
    Yb[i] = f2bf(v);
  }
  __syncthreads();
  v8f a0 = {}, a1 = {};
#pragma unroll
  for (int kc = 0; kc < 4; ++kc) {
    v16bf a = load_a(Yb, HDIM, kc, lane);
    a0 = wmma_bf16(a, load_b(outf, 8, kc, 2 * wv + 0, lane), a0);
    a1 = wmma_bf16(a, load_b(outf, 8, kc, 2 * wv + 1, lane), a1);
  }
  int kh = lane >> 4, nl = lane & 15;
#pragma unroll
  for (int t = 0; t < 2; ++t) {
    int n = (2 * wv + t) * 16 + nl; v8f acc = t ? a1 : a0;
#pragma unroll
    for (int r = 0; r < 8; ++r) {
      int node = n0 + r + 8 * kh;
      if (node < Ntot) out[(size_t)node * HDIM + n] = acc[r];
    }
  }
}

// ---------------- host launcher ----------------
extern "C" void kernel_launch(void* const* d_in, const int* in_sizes, int n_in,
                              void* d_out, int out_size, void* d_ws, size_t ws_size,
                              hipStream_t stream) {
  const float* h        = (const float*)d_in[0];
  const int*   eidx     = (const int*)  d_in[1];
  const float* eattr    = (const float*)d_in[2];
  const float* e_w1     = (const float*)d_in[3];
  const float* e_b1     = (const float*)d_in[4];
  const float* e_w2     = (const float*)d_in[5];
  const float* e_b2     = (const float*)d_in[6];
  const float* inf_w    = (const float*)d_in[7];
  const float* inf_b    = (const float*)d_in[8];
  const float* n_w1     = (const float*)d_in[9];
  const float* n_b1     = (const float*)d_in[10];
  const float* n_w2     = (const float*)d_in[11];
  const float* n_b2     = (const float*)d_in[12];
  const float* in_proj  = (const float*)d_in[13];
  const float* conv_w   = (const float*)d_in[14];
  const float* conv_b   = (const float*)d_in[15];
  const float* x_proj   = (const float*)d_in[16];
  const float* dt_w     = (const float*)d_in[17];
  const float* dt_b     = (const float*)d_in[18];
  const float* A_log    = (const float*)d_in[19];
  const float* Dv       = (const float*)d_in[20];
  const float* out_proj = (const float*)d_in[21];
  float* out = (float*)d_out;

  int Ntot = in_sizes[0] / HDIM;
  int Etot = in_sizes[1] / 2;
  int NC   = (Ntot + CH_L - 1) / CH_L;

  char* base = (char*)d_ws;
  size_t off = 0;
  auto alloc = [&](size_t bytes) -> char* {
    char* r = base + off;
    off = (off + bytes + 255) & ~(size_t)255;
    return r;
  };
  unsigned short* hb    = (unsigned short*)alloc((size_t)Ntot * HDIM * 2);
  unsigned short* w1f   = (unsigned short*)alloc((size_t)9 * 8 * 512 * 2);
  unsigned short* nw1f  = (unsigned short*)alloc((size_t)8 * 8 * 512 * 2);
  unsigned short* w2f   = (unsigned short*)alloc((size_t)4 * 8 * 512 * 2);
  unsigned short* nw2f  = (unsigned short*)alloc((size_t)4 * 8 * 512 * 2);
  unsigned short* outf  = (unsigned short*)alloc((size_t)4 * 8 * 512 * 2);
  unsigned short* inpf  = (unsigned short*)alloc((size_t)4 * 16 * 512 * 2);
  unsigned short* xpf   = (unsigned short*)alloc((size_t)4 * 9 * 512 * 2);
  unsigned short* dtwf  = (unsigned short*)alloc((size_t)1 * 8 * 512 * 2);
  float* Aneg   = (float*)alloc((size_t)HDIM * DSQ * 4);
  float* mi     = (float*)alloc((size_t)Ntot * HDIM * 4);
  float* agg    = (float*)alloc((size_t)Ntot * HDIM * 4);
  float* xraw   = (float*)alloc((size_t)Ntot * HDIM * 4);
  float* zbuf   = (float*)alloc((size_t)Ntot * HDIM * 4);
  float* xcv    = (float*)alloc((size_t)Ntot * HDIM * 4);
  float* Bm     = (float*)alloc((size_t)Ntot * DSQ * 4);
  float* Cm     = (float*)alloc((size_t)Ntot * DSQ * 4);
  float* delta  = (float*)alloc((size_t)Ntot * HDIM * 4);
  float* ybuf   = (float*)alloc((size_t)Ntot * HDIM * 4);
  float* dsum   = (float*)alloc((size_t)NC * HDIM * 4);
  float* Slocal = (float*)alloc((size_t)NC * HDIM * DSQ * 4);
  float* Sinit  = (float*)alloc((size_t)NC * HDIM * DSQ * 4);

  hipMemsetAsync(mi,  0, (size_t)Ntot * HDIM * 4, stream);
  hipMemsetAsync(agg, 0, (size_t)Ntot * HDIM * 4, stream);

  prep_frags<<<64, 256, 0, stream>>>(e_w1,     w1f,  9, 8,  258, 128);
  prep_frags<<<64, 256, 0, stream>>>(n_w1,     nw1f, 8, 8,  256, 128);
  prep_frags<<<64, 256, 0, stream>>>(e_w2,     w2f,  4, 8,  128, 128);
  prep_frags<<<64, 256, 0, stream>>>(n_w2,     nw2f, 4, 8,  128, 128);
  prep_frags<<<64, 256, 0, stream>>>(out_proj, outf, 4, 8,  128, 128);
  prep_frags<<<64, 256, 0, stream>>>(in_proj,  inpf, 4, 16, 128, 256);
  prep_frags<<<64, 256, 0, stream>>>(x_proj,   xpf,  4, 9,  128, 136);
  prep_frags<<<64, 256, 0, stream>>>(dt_w,     dtwf, 1, 8,  8,   128);
  prep_misc <<<256, 256, 0, stream>>>(h, A_log, hb, Aneg, Ntot * HDIM, HDIM * DSQ);

  k_edge<<<(Etot + 15) / 16, 128, 0, stream>>>(hb, eidx, eattr, w1f, e_b1, w2f, e_b2,
                                               inf_w, inf_b, mi, agg, Etot, Ntot);
  k_node<<<(Ntot + 15) / 16, 128, 0, stream>>>(hb, h, agg, mi, nw1f, n_b1, nw2f, n_b2,
                                               inpf, xraw, zbuf, Ntot);
  k_conv<<<(Ntot + 15) / 16, 128, 0, stream>>>(xraw, conv_w, conv_b, xpf, dtwf, dt_b,
                                               xcv, Bm, Cm, delta, Ntot);
  k_scan1<<<NC, 256, 0, stream>>>(delta, xcv, Bm, Aneg, Slocal, dsum, Ntot, CH_L);
  k_scan2<<<1, 256, 0, stream>>>(Slocal, dsum, Aneg, Sinit, NC);
  k_scan3<<<NC, 256, 0, stream>>>(delta, xcv, Bm, Cm, Aneg, Sinit, ybuf, Ntot, CH_L);
  k_final<<<(Ntot + 15) / 16, 128, 0, stream>>>(ybuf, xcv, zbuf, Dv, outf, out, Ntot);
  (void)n_in; (void)out_size; (void)ws_size;
}
